// TrajectoryAgent_63410897158575
// MI455X (gfx1250) — matrix operand order repaired and checked by reading, not compiled, not run on hardware
//
#include <hip/hip_runtime.h>
#include <math.h>

typedef __attribute__((ext_vector_type(16))) __bf16 v16bf;
typedef __attribute__((ext_vector_type(8)))  __bf16 v8bf;
typedef __attribute__((ext_vector_type(8)))  float  v8f;
typedef unsigned short u16;

#define TOKD 128          // token dim D
#define ACTD 256          // action dim A = 2H
#define MT   128          // edges per workgroup (M-tile)
#define SA_STRIDE   136   // 128 + 8 halves padding (bank-conflict avoidance, keeps 16B align)
#define SW_STRIDE_D 136
#define SW_STRIDE_A 264   // 256 + 8 halves
#define SG_STRIDE   264

// LDS partition (bytes)
#define OFF_SA   0
#define SZ_SA    (MT * SA_STRIDE * 2)            // 34816
#define OFF_SW   (OFF_SA + SZ_SA)
#define SZ_SW    (ACTD * SW_STRIDE_A * 2)        // 135168
#define OFF_SG   (OFF_SW + SZ_SW)
#define SZ_SG    (MT * SG_STRIDE * 2)            // 67584
#define OFF_SACT (OFF_SG + SZ_SG)
#define SZ_SACT  (MT * SG_STRIDE * 2)            // 67584
#define SMEM_TOTAL (OFF_SACT + SZ_SACT)          // 305152 bytes < 320KB/WGP

__device__ __forceinline__ u16 f2bf(float f) {   // fp32 -> bf16, round-to-nearest-even
  unsigned x = __float_as_uint(f);
  x += 0x7FFFu + ((x >> 16) & 1u);
  return (u16)(x >> 16);
}
__device__ __forceinline__ float bf2f(u16 h) {
  return __uint_as_float(((unsigned)h) << 16);
}
// sigmoid with single-instruction v_rcp_f32 (arg >= 1, safe; gate is bf16-quantized anyway)
__device__ __forceinline__ float fast_sigmoid(float x) {
  return __builtin_amdgcn_rcpf(1.0f + __expf(-x));
}

union V16U { v16bf v; v8bf h[2]; };

// One 16x16 C tile of  C = A(MT x K) * B  with B staged transposed ([N][K]) in LDS.
// A-frag per ISA: lane<16 holds M=lane, K = {0..7, 16..23}; lane>=16 M=lane-16, K={8..15, 24..31}.
// B-frag per ISA: lane<16 holds N=lane, K = {0..7, 16..23}; lane>=16 N=lane-16, K={8..15, 24..31}.
__device__ __forceinline__ v8f gemm_frag(const u16* __restrict__ sAb, const u16* __restrict__ sBb,
                                         int strideA, int strideB, int nk, int mt, int nt) {
  const int lane = threadIdx.x & 31;
  const int hsel = lane >> 4;
  const int l15  = lane & 15;
  const u16* ap = sAb + (mt * 16 + l15) * strideA + hsel * 8;
  const u16* bp = sBb + (nt * 16 + l15) * strideB + hsel * 8;
  v8f c = {};
#pragma unroll
  for (int kc = 0; kc < nk; ++kc) {
    V16U a, b;
    a.h[0] = *(const v8bf*)(const void*)(ap + kc * 32);
    a.h[1] = *(const v8bf*)(const void*)(ap + kc * 32 + 16);
    b.h[0] = *(const v8bf*)(const void*)(bp + kc * 32);
    b.h[1] = *(const v8bf*)(const void*)(bp + kc * 32 + 16);
    c = __builtin_amdgcn_wmma_f32_16x16x32_bf16(false, a.v, false, b.v, (short)0, c,
                                                false, false);
  }
  return c;
}

// Stage W[K][N] (row-major f32, N = 2^nshift) transposed into LDS as bf16 [N][K] w/ padded stride.
__device__ __forceinline__ void stage_wT(u16* dst, const float* __restrict__ W,
                                         int K, int nshift, int stride) {
  const int total = K << nshift;
  const int nmask = (1 << nshift) - 1;
  for (int i = threadIdx.x; i < total; i += 256) {
    int k = i >> nshift;
    int n = i & nmask;
    dst[n * stride + k] = f2bf(W[i]);
  }
}

// Stage a [MT x 128] f32 token tile into LDS as bf16 (MODE 0: direct rows from `tok`;
// MODE 1: gather rows tok[idx[E+e]] (tail nodes); MODE 2: gather rows tok[idx[e]] (batch)).
template <int MODE>
__device__ __forceinline__ void stage_tokens(u16* dst, const float* __restrict__ tok,
                                             const int* __restrict__ idx, int base, int E) {
  const int tid = threadIdx.x;
  const int cg  = (tid & 31) * 4;               // float column group (16B)
#pragma unroll 4
  for (int it = 0; it < 16; ++it) {
    int row = (tid >> 5) + it * 8;
    int e = base + row;
    if (e >= E) e = E - 1;                      // clamp (keeps gathers in-bounds)
    size_t srow;
    if (MODE == 0)      srow = (size_t)e;
    else if (MODE == 1) srow = (size_t)idx[(size_t)E + e];
    else                srow = (size_t)idx[e];
    float4 v = *(const float4*)(tok + srow * TOKD + cg);
    ushort4 o;
    o.x = f2bf(v.x); o.y = f2bf(v.y); o.z = f2bf(v.z); o.w = f2bf(v.w);
    *(ushort4*)(dst + row * SA_STRIDE + cg) = o;
  }
}

extern "C" __global__ __launch_bounds__(256)
void traj_agent_kernel(const float* __restrict__ rel,   const float* __restrict__ nodes,
                       const int*   __restrict__ eidx,  const float* __restrict__ qtok,
                       const int*   __restrict__ ebatch,
                       const float* __restrict__ Wrel,  const float* __restrict__ Wnode,
                       const float* __restrict__ Wq,    const float* __restrict__ Wm1,
                       const float* __restrict__ b1,    const float* __restrict__ Wm2,
                       const float* __restrict__ b2,    float* __restrict__ out, int E) {
  extern __shared__ char smem[];
  u16* sA   = (u16*)(smem + OFF_SA);    // input token tile, bf16 [128][136]
  u16* sW   = (u16*)(smem + OFF_SW);    // current weight^T,  bf16 [N][Kpad]
  u16* sG   = (u16*)(smem + OFF_SG);    // gate tile bf16 [128][264]; reused as h after phase 3
  u16* sAct = (u16*)(smem + OFF_SACT);  // action_key tile bf16 [128][264]
  u16* sH   = sG;

  const int base = blockIdx.x * MT;
  const int tid  = threadIdx.x;
  const int wave = tid >> 5;
  const int lane = tid & 31;
  const int l15  = lane & 15;
  const int hsel = lane >> 4;

  // Prefetch the relation-token tile needed in phase 1 (emits global_prefetch_b8).
  __builtin_prefetch(rel + (size_t)base * TOKD + (size_t)tid * 32, 0, 0);
  __builtin_prefetch(rel + (size_t)base * TOKD + 8192 + (size_t)tid * 32, 0, 0);

  // ---------- Phase 0: gate = sigmoid(q_tokens[edge_batch] @ W_qgate), [128 x 256] ----------
  stage_tokens<2>(sA, qtok, ebatch, base, E);
  stage_wT(sW, Wq, TOKD, 8, SW_STRIDE_D);
  __syncthreads();
#pragma unroll 2
  for (int i = 0; i < 16; ++i) {                       // 128 tiles / 8 waves
    int t = i * 8 + wave, mt = t & 7, nt = t >> 3;
    v8f c = gemm_frag(sA, sW, SA_STRIDE, SW_STRIDE_D, 4, mt, nt);
    int Nc = nt * 16 + l15, Mb = mt * 16 + hsel * 8;
#pragma unroll
    for (int r = 0; r < 8; ++r) {
      float g = fast_sigmoid(c[r]);
      sG[(Mb + r) * SG_STRIDE + Nc] = f2bf(g);
    }
  }
  __syncthreads();

  // ---------- Phase 1: act[:, :128] = (rel_tokens @ W_rel) * gate[:, :128] ----------
  stage_tokens<0>(sA, rel, eidx, base, E);
  stage_wT(sW, Wrel, TOKD, 7, SW_STRIDE_D);
  __syncthreads();
#pragma unroll 2
  for (int i = 0; i < 8; ++i) {                        // 64 tiles / 8 waves
    int t = i * 8 + wave, mt = t & 7, nt = t >> 3;
    v8f c = gemm_frag(sA, sW, SA_STRIDE, SW_STRIDE_D, 4, mt, nt);
    int Nc = nt * 16 + l15, Mb = mt * 16 + hsel * 8;
#pragma unroll
    for (int r = 0; r < 8; ++r) {
      float g = bf2f(sG[(Mb + r) * SG_STRIDE + Nc]);
      sAct[(Mb + r) * SG_STRIDE + Nc] = f2bf(c[r] * g);
    }
  }
  __syncthreads();

  // ---------- Phase 2: act[:, 128:] = (node_tokens[tail] @ W_node) * gate[:, 128:] ----------
  stage_tokens<1>(sA, nodes, eidx, base, E);
  stage_wT(sW, Wnode, TOKD, 7, SW_STRIDE_D);
  __syncthreads();
#pragma unroll 2
  for (int i = 0; i < 8; ++i) {
    int t = i * 8 + wave, mt = t & 7, nt = t >> 3;
    v8f c = gemm_frag(sA, sW, SA_STRIDE, SW_STRIDE_D, 4, mt, nt);
    int Nc = nt * 16 + l15 + 128, Mb = mt * 16 + hsel * 8;
#pragma unroll
    for (int r = 0; r < 8; ++r) {
      float g = bf2f(sG[(Mb + r) * SG_STRIDE + Nc]);
      sAct[(Mb + r) * SG_STRIDE + Nc] = f2bf(c[r] * g);
    }
  }
  __syncthreads();

  // ---------- Phase 3: h = gelu_exact(act @ W_mlp1 + b1), [128 x 256] ----------
  stage_wT(sW, Wm1, ACTD, 8, SW_STRIDE_A);
  __syncthreads();
  for (int i = 0; i < 16; ++i) {
    int t = i * 8 + wave, mt = t & 7, nt = t >> 3;
    v8f c = gemm_frag(sAct, sW, SG_STRIDE, SW_STRIDE_A, 8, mt, nt);
    int Nc = nt * 16 + l15, Mb = mt * 16 + hsel * 8;
    float bias = b1[Nc];
#pragma unroll
    for (int r = 0; r < 8; ++r) {
      float x = c[r] + bias;
      float gl = 0.5f * x * (1.0f + erff(x * 0.70710678118654752f));
      sH[(Mb + r) * SG_STRIDE + Nc] = f2bf(gl);        // sH aliases sG (gate now dead)
    }
  }
  __syncthreads();

  // ---------- Phase 4: out = h @ W_mlp2 + b2 ----------
  stage_wT(sW, Wm2, ACTD, 8, SW_STRIDE_A);
  __syncthreads();
  for (int i = 0; i < 16; ++i) {
    int t = i * 8 + wave, mt = t & 7, nt = t >> 3;
    v8f c = gemm_frag(sH, sW, SG_STRIDE, SW_STRIDE_A, 8, mt, nt);
    int Nc = nt * 16 + l15, Mb = mt * 16 + hsel * 8;
    float bias = b2[Nc];
#pragma unroll
    for (int r = 0; r < 8; ++r) {
      int e = base + Mb + r;
      if (e < E) out[(size_t)e * ACTD + Nc] = c[r] + bias;
    }
  }
}

extern "C" void kernel_launch(void* const* d_in, const int* in_sizes, int n_in,
                              void* d_out, int out_size, void* d_ws, size_t ws_size,
                              hipStream_t stream) {
  const float* rel    = (const float*)d_in[0];
  const float* nodes  = (const float*)d_in[1];
  const int*   eidx   = (const int*)  d_in[2];
  const float* qtok   = (const float*)d_in[3];
  const int*   ebatch = (const int*)  d_in[4];
  const float* Wrel   = (const float*)d_in[5];
  const float* Wnode  = (const float*)d_in[6];
  const float* Wq     = (const float*)d_in[7];
  const float* Wm1    = (const float*)d_in[8];
  const float* b1     = (const float*)d_in[9];
  const float* Wm2    = (const float*)d_in[10];
  const float* b2     = (const float*)d_in[11];
  float* out = (float*)d_out;

  const int E = in_sizes[4];                 // edge_batch has one entry per edge
  const int grid = (E + MT - 1) / MT;

  (void)hipFuncSetAttribute((const void*)traj_agent_kernel,
                            hipFuncAttributeMaxDynamicSharedMemorySize, SMEM_TOTAL);
  hipLaunchKernelGGL(traj_agent_kernel, dim3(grid), dim3(256), SMEM_TOTAL, stream,
                     rel, nodes, eidx, qtok, ebatch, Wrel, Wnode, Wq, Wm1, b1, Wm2, b2,
                     out, E);
}
